// ExplicitConv2dLayer_37125697307139
// MI455X (gfx1250) — matrix operand order repaired and checked by reading, not compiled, not run on hardware
//
#include <hip/hip_runtime.h>
#include <hip/hip_bf16.h>
#include <stddef.h>

typedef __attribute__((ext_vector_type(16))) _Float16 v16h;
typedef __attribute__((ext_vector_type(8)))  float    v8f;

// Problem constants
#define BATCH 8
#define CIN   64
#define COUT  64
#define HH    256
#define WW    256
#define KK    3
// Interior weight index (gap=1): (0+1)*3+(0+1) = 4
#define IDX_INT 4
#define WSET   (COUT*CIN*KK*KK)   // 36864 elems per weight set

// Main-kernel tiling: 8 rows x 32 cols per workgroup
//   halo read-amplification = (10/8)*(34/32) ~= 1.33x  (was 1.55x)
#define TILE_J   32
#define TILE_R   8
#define LDS_COLS (TILE_J + 2)       // 34 (halo)
#define LDS_ROWS (TILE_R + 2)       // 10 (halo)
#define CI_PAD   72                 // per-column ci stride in halves (144B: 16B aligned, bank-friendly)

// -------------------------------------------------------------------------
// Kernel 1: repack interior weight set (f32) into WMMA A-fragment layout (f16)
// Fragment index s = (kh*3+kw)*2 + h  (h = ci half), then coTile(4), lane(32), e(16).
// A 16x32 f16 layout (contiguous, matches B): lane<16 -> K in {0..15}; lane>=16 -> {16..31}.
__global__ void prep_weights_kernel(const float* __restrict__ Wall,
                                    _Float16* __restrict__ wsA) {
    int t = blockIdx.x * 256 + threadIdx.x;
    if (t >= 18 * 4 * 32 * 16) return;
    int e      =  t        & 15;
    int lane   = (t >> 4)  & 31;
    int coTile = (t >> 9)  & 3;
    int s      =  t >> 11;          // 0..17
    int khkw   = s >> 1;            // 0..8
    int h      = s & 1;
    int kh     = khkw / 3;
    int kw     = khkw % 3;
    int sel    = lane >> 4;
    int m      = lane & 15;
    int kk     = e + sel * 16;   // K index within 32-chunk
    int ci     = 32 * h + kk;
    int co     = coTile * 16 + m;
    const float* W4 = Wall + (size_t)IDX_INT * WSET;
    float v = W4[(size_t)co * (CIN * KK * KK) + ci * (KK * KK) + kh * KK + kw];
    wsA[t] = (_Float16)v;
}

// -------------------------------------------------------------------------
// Kernel 2: interior implicit-GEMM conv via v_wmma_f32_16x16x32_f16.
// Grid: 2048 blocks = 8 batches * 32 row-blocks * 8 j-blocks; 512 threads (16 waves).
__global__ void __launch_bounds__(512)
conv_wmma_kernel(const float* __restrict__ img,
                 const _Float16* __restrict__ wsA,
                 const float* __restrict__ Ball,
                 float* __restrict__ out) {
    __shared__ _Float16 lds[LDS_ROWS * LDS_COLS * CI_PAD];   // 48,960 B

    const int tid  = threadIdx.x;
    const int wg   = blockIdx.x;
    const int bIdx = wg >> 8;              // batch  (256 WGs per batch)
    const int rem  = wg & 255;
    const int i0   = (rem >> 3) * TILE_R;  // 32 row-blocks of 8
    const int j0   = (rem & 7) * TILE_J;   // 8  col-blocks of 32

    // ---- stage 10 input rows x 34 cols x 64 ci into LDS, f32 -> f16, [col][ci] ----
    const float* imgB = img + ((size_t)bIdx * CIN) * (HH * WW);
    const int NSTAGE = LDS_ROWS * LDS_COLS * CIN;   // 21760
    for (int e = tid; e < NSTAGE; e += 512) {
        int r   = e / (LDS_COLS * CIN);
        int r2  = e % (LDS_COLS * CIN);
        int ci  = r2 / LDS_COLS;
        int c   = r2 % LDS_COLS;              // consecutive tid -> consecutive col (coalesced)
        int gi  = i0 - 1 + r;
        int gc  = j0 - 1 + c;
        float v = 0.0f;
        if ((unsigned)gi < (unsigned)HH && (unsigned)gc < (unsigned)WW)
            v = imgB[((size_t)ci << 16) + (gi << 8) + gc];
        lds[(r * LDS_COLS + c) * CI_PAD + ci] = (_Float16)v;
    }
    __syncthreads();

    // ---- per-wave 16(co) x 16(j) x 4(rows) output tile ----
    const int wave   = tid >> 5;
    const int lane   = tid & 31;
    const int coTile = wave & 3;             // 4 co tiles of 16
    const int jBase  = ((wave >> 2) & 1) * 16;  // 2 j sub-tiles of 16
    const int r0     = (wave >> 3) * 4;      // 2 row groups of 4
    const int sel    = lane >> 4;
    const int m      = lane & 15;

    v8f acc[4] = {};

    #pragma unroll
    for (int s = 0; s < 18; ++s) {       // K = 576 in 18 steps of 32
        const int khkw = s >> 1;
        const int h    = s & 1;
        const int kh   = khkw / 3;
        const int kw   = khkw % 3;

        // A fragment: 32B contiguous per lane from pre-packed weights (L2-resident)
        v16h a = *(const v16h*)(wsA + ((size_t)(s * 4 + coTile) * 32 + lane) * 16);

        const int c    = jBase + m + kw;          // 0..33
        const int boff = 32 * h + 16 * sel;       // ci start within column

        #pragma unroll
        for (int r = 0; r < 4; ++r) {
            const int rr = r0 + r + kh;           // 0..9
            v16h bfrag = *(const v16h*)(&lds[(rr * LDS_COLS + c) * CI_PAD + boff]);
            acc[r] = __builtin_amdgcn_wmma_f32_16x16x32_f16(
                false, a, false, bfrag, (short)0, acc[r], false, false);
        }
    }

    // ---- bias + store (C layout: lane&15 = N=j, VGPR v = M row, lane>=16 -> M+8) ----
    const float* bias4 = Ball + IDX_INT * COUT;
    const int jj = j0 + jBase + m;
    const size_t outBase = ((size_t)bIdx * COUT) * (HH * WW);
    #pragma unroll
    for (int r = 0; r < 4; ++r) {
        const int i = i0 + r0 + r;
        #pragma unroll
        for (int v = 0; v < 8; ++v) {
            const int co = coTile * 16 + sel * 8 + v;
            out[outBase + (size_t)co * (HH * WW) + i * WW + jj] = acc[r][v] + bias4[co];
        }
    }
}

// -------------------------------------------------------------------------
// Kernel 3: exact f32 recompute of the border pixels with per-pixel weight set.
__global__ void border_fixup_kernel(const float* __restrict__ img,
                                    const float* __restrict__ Wall,
                                    const float* __restrict__ Ball,
                                    float* __restrict__ out) {
    const int PX = 2 * WW + 2 * (HH - 2);   // 1020 border pixels
    int t = blockIdx.x * 256 + threadIdx.x;
    if (t >= BATCH * PX * COUT) return;
    int co = t & 63;
    int p  = (t >> 6) % PX;
    int b  = t / (COUT * PX);
    int i, j;
    if (p < WW)            { i = 0;      j = p; }
    else if (p < 2 * WW)   { i = HH - 1; j = p - WW; }
    else if (p < 2 * WW + (HH - 2)) { j = 0;      i = 1 + (p - 2 * WW); }
    else                   { j = WW - 1; i = 1 + (p - (2 * WW + (HH - 2))); }

    int dh = (i == 0) ? 1 : ((i == HH - 1) ? -1 : 0);
    int dw = (j == 0) ? 1 : ((j == WW - 1) ? -1 : 0);
    int idx = (dh + 1) * 3 + (dw + 1);

    const float* Wp = Wall + (size_t)idx * WSET + (size_t)co * (CIN * KK * KK);
    float sum = Ball[idx * COUT + co];
    const float* imgB = img + ((size_t)b * CIN) * (HH * WW);
    for (int kh = 0; kh < KK; ++kh) {
        int gi = i + kh - 1;
        if ((unsigned)gi >= (unsigned)HH) continue;
        for (int kw = 0; kw < KK; ++kw) {
            int gj = j + kw - 1;
            if ((unsigned)gj >= (unsigned)WW) continue;
            const float* ip = imgB + gi * WW + gj;
            const float* wp = Wp + kh * KK + kw;
            #pragma unroll 8
            for (int ci = 0; ci < CIN; ++ci)
                sum += ip[(size_t)ci * (HH * WW)] * wp[ci * (KK * KK)];
        }
    }
    out[(((size_t)b * COUT + co) * HH + i) * WW + j] = sum;
}

// -------------------------------------------------------------------------
extern "C" void kernel_launch(void* const* d_in, const int* in_sizes, int n_in,
                              void* d_out, int out_size, void* d_ws, size_t ws_size,
                              hipStream_t stream) {
    const float* img  = (const float*)d_in[0];   // [8,64,256,256] f32
    const float* Wall = (const float*)d_in[1];   // [9,64,64,3,3]  f32
    const float* Ball = (const float*)d_in[2];   // [9,64]         f32
    float* out = (float*)d_out;
    _Float16* wsA = (_Float16*)d_ws;             // 36864 halves = 73,728 B

    // 1) pack interior weights into WMMA A-fragment f16 layout
    prep_weights_kernel<<<144, 256, 0, stream>>>(Wall, wsA);

    // 2) interior conv via WMMA: 8 batches * 32 row-blocks * 8 j-blocks = 2048 WGs
    conv_wmma_kernel<<<BATCH * 32 * 8, 512, 0, stream>>>(img, wsA, Ball, out);

    // 3) overwrite border pixels with exact per-idx conv
    border_fixup_kernel<<<(BATCH * 1020 * COUT) / 256, 256, 0, stream>>>(img, Wall, Ball, out);
}
